// ParallelMemoryModule_18674517803206
// MI455X (gfx1250) — matrix operand order repaired and checked by reading, not compiled, not run on hardware
//
#include <hip/hip_runtime.h>
#include <math.h>

// ---------------- types for WMMA fragments (gfx1250, wave32) ----------------
typedef __bf16  bf16x8  __attribute__((ext_vector_type(8)));
typedef __bf16  bf16x16 __attribute__((ext_vector_type(16)));
typedef float   f32x8   __attribute__((ext_vector_type(8)));

union Frag { bf16x16 v; bf16x8 h[2]; };

// Problem constants (match reference)
#define S_  2048
#define B_  2
#define M_  4096
#define H_  2048
#define NH_ 16
#define D_  128
#define FFN_ 8192
#define ROWS_   (S_ * B_)   // 4096
#define KVROWS_ (M_ * B_)   // 8192

__device__ __forceinline__ float gelu_exact(float x) {
    return 0.5f * x * (1.0f + erff(x * 0.70710678118654752f));
}

// Async copy 16B (8 bf16) per lane from global to LDS, tracked by ASYNCcnt.
// CDNA5 ISA: GLOBAL_LOAD_ASYNC_TO_LDS_B128, GV mode.
__device__ __forceinline__ void async_copy_b128(const void* gptr, unsigned ldsOff) {
    asm volatile("global_load_async_to_lds_b128 %0, %1, off"
                 :: "v"(ldsOff), "v"(gptr) : "memory");
}
__device__ __forceinline__ void wait_async0() {
    asm volatile("s_wait_asynccnt 0x0" ::: "memory");
}

// ---------------- f32 -> bf16 elementwise convert ----------------
__global__ void cvt_f32_bf16(const float* __restrict__ in, __bf16* __restrict__ out, size_t n) {
    size_t i = (size_t)blockIdx.x * blockDim.x + threadIdx.x;
    size_t stride = (size_t)gridDim.x * blockDim.x;
    for (; i < n; i += stride) out[i] = (__bf16)in[i];
}

// ---------------- f32 [K,N] -> bf16 transposed [N,K] (LDS tiled) ----------------
__global__ __launch_bounds__(256) void cvt_transpose_bf16(
    const float* __restrict__ in, __bf16* __restrict__ out, int K, int N) {
    __shared__ float tile[32][33];
    int k0 = blockIdx.y * 32, n0 = blockIdx.x * 32;
    for (int i = threadIdx.y; i < 32; i += 8) {
        tile[i][threadIdx.x] = in[(size_t)(k0 + i) * N + n0 + threadIdx.x];
    }
    __syncthreads();
    for (int i = threadIdx.y; i < 32; i += 8) {
        out[(size_t)(n0 + i) * K + k0 + threadIdx.x] = (__bf16)tile[threadIdx.x][i];
    }
}

// ---------------- layernorm over H=2048, writes bf16 ----------------
__global__ __launch_bounds__(256) void layernorm_bf16(
    const float* __restrict__ x, const float* __restrict__ g, const float* __restrict__ bta,
    __bf16* __restrict__ out) {
    int row = blockIdx.x;
    const float* xr = x + (size_t)row * H_;
    float v[8];
    float s = 0.f, s2 = 0.f;
#pragma unroll
    for (int t = 0; t < 8; ++t) {
        v[t] = xr[threadIdx.x + t * 256];
        s += v[t]; s2 += v[t] * v[t];
    }
#pragma unroll
    for (int m = 1; m < 32; m <<= 1) { s += __shfl_xor(s, m); s2 += __shfl_xor(s2, m); }
    __shared__ float red[2][8];
    int wid = threadIdx.x >> 5, lane = threadIdx.x & 31;
    if (lane == 0) { red[0][wid] = s; red[1][wid] = s2; }
    __syncthreads();
    if (wid == 0) {
        s  = (lane < 8) ? red[0][lane] : 0.f;
        s2 = (lane < 8) ? red[1][lane] : 0.f;
#pragma unroll
        for (int m = 1; m < 8; m <<= 1) { s += __shfl_xor(s, m); s2 += __shfl_xor(s2, m); }
        if (lane == 0) { red[0][0] = s; red[1][0] = s2; }
    }
    __syncthreads();
    s = red[0][0]; s2 = red[1][0];
    float mean = s * (1.0f / H_);
    float var  = s2 * (1.0f / H_) - mean * mean;
    float inv  = rsqrtf(var + 1e-5f);
    __bf16* orow = out + (size_t)row * H_;
#pragma unroll
    for (int t = 0; t < 8; ++t) {
        int c = threadIdx.x + t * 256;
        orow[c] = (__bf16)((v[t] - mean) * inv * g[c] + bta[c]);
    }
}

// ---------------- WMMA GEMM with async LDS double-buffering ----------------
// C = epilogue(A[R,K] @ Wt[N,K]^T + bias)
// Block: 256 thr (8 waves), block tile 128x128, wave tile 64x32 (4x2 WMMA accums).
// Per K-step (32): stage A[128x32] and B[128x32] bf16 tiles into LDS via
// global_load_async_to_lds_b128 (ASYNCcnt), double-buffered so the next tile
// streams in while WMMAs consume the current one.
__global__ __launch_bounds__(256) void gemm_wmma(
    const __bf16* __restrict__ A,    // [R,K] row-major bf16
    const __bf16* __restrict__ Wt,   // [N,K] row-major bf16 (W transposed)
    const float*  __restrict__ bias, // [N] or null
    const float*  __restrict__ resid,// [R,N] or null
    float*        __restrict__ Cf,   // [R,N] f32 or null
    __bf16*       __restrict__ Cb,   // [R,N] bf16 or null
    int R, int N, int K, int gelu) {
    __shared__ __align__(16) __bf16 Abuf[2][128][32];
    __shared__ __align__(16) __bf16 Bbuf[2][128][32];

    int lane = threadIdx.x & 31;
    int wid  = threadIdx.x >> 5;
    int r16  = lane & 15;
    int half = lane >> 4;
    int rows0 = blockIdx.y * 128;
    int cols0 = blockIdx.x * 128;
    int wrow = wid & 1;        // 2 row groups of 64
    int wcol = wid >> 1;       // 4 col groups of 32

    // staging assignment: thread t copies row t>>1, k-half t&1 (16 bf16 = 2x b128)
    int srow = threadIdx.x >> 1;
    int sseg = threadIdx.x & 1;

    int NK = K >> 5;
    // prologue: stage k-step 0 into buffer 0
    {
        const __bf16* ga = A  + (size_t)(rows0 + srow) * K + sseg * 16;
        const __bf16* gb = Wt + (size_t)(cols0 + srow) * K + sseg * 16;
        unsigned la = (unsigned)(uintptr_t)(&Abuf[0][srow][sseg * 16]);
        unsigned lb = (unsigned)(uintptr_t)(&Bbuf[0][srow][sseg * 16]);
        async_copy_b128(ga,     la);
        async_copy_b128(ga + 8, la + 16);
        async_copy_b128(gb,     lb);
        async_copy_b128(gb + 8, lb + 16);
    }

    f32x8 acc[4][2] = {};
    for (int kb = 0; kb < NK; ++kb) {
        wait_async0();         // my staged transfers for step kb landed
        __syncthreads();       // everyone's transfers landed; prev compute done
        if (kb + 1 < NK) {     // stream next tile into the other buffer
            int k0n = (kb + 1) << 5;
            int bn  = (kb + 1) & 1;
            const __bf16* ga = A  + (size_t)(rows0 + srow) * K + k0n + sseg * 16;
            const __bf16* gb = Wt + (size_t)(cols0 + srow) * K + k0n + sseg * 16;
            unsigned la = (unsigned)(uintptr_t)(&Abuf[bn][srow][sseg * 16]);
            unsigned lb = (unsigned)(uintptr_t)(&Bbuf[bn][srow][sseg * 16]);
            async_copy_b128(ga,     la);
            async_copy_b128(ga + 8, la + 16);
            async_copy_b128(gb,     lb);
            async_copy_b128(gb + 8, lb + 16);
        }
        int buf = kb & 1;
        Frag a[4], b[2];
#pragma unroll
        for (int i = 0; i < 4; ++i) {
            const __bf16* ap = Abuf[buf][wrow * 64 + i * 16 + r16];
            a[i].h[0] = *(const bf16x8*)(ap + half * 8);
            a[i].h[1] = *(const bf16x8*)(ap + 16 + half * 8);
        }
#pragma unroll
        for (int j = 0; j < 2; ++j) {
            const __bf16* bp = Bbuf[buf][wcol * 32 + j * 16 + r16] + half * 8;
            b[j].h[0] = *(const bf16x8*)(bp);
            b[j].h[1] = *(const bf16x8*)(bp + 16);
        }
#pragma unroll
        for (int i = 0; i < 4; ++i)
#pragma unroll
            for (int j = 0; j < 2; ++j)
                acc[i][j] = __builtin_amdgcn_wmma_f32_16x16x32_bf16(
                    false, a[i].v, false, b[j].v, (short)0, acc[i][j], false, false);
    }
    // epilogue
#pragma unroll
    for (int i = 0; i < 4; ++i) {
#pragma unroll
        for (int j = 0; j < 2; ++j) {
            int cc = cols0 + wcol * 32 + j * 16 + r16;
            float bv = bias ? bias[cc] : 0.f;
#pragma unroll
            for (int e = 0; e < 8; ++e) {
                int rr = rows0 + wrow * 64 + i * 16 + e + 8 * half;
                float val = acc[i][j][e] + bv;
                if (gelu) val = gelu_exact(val);
                size_t idx = (size_t)rr * N + cc;
                if (resid) val += resid[idx];
                if (Cf) Cf[idx] = val;
                if (Cb) Cb[idx] = (__bf16)val;
            }
        }
    }
}

// ---------------- L2-normalize q: qraw[S*B, H] f32 -> q_n[BH, S, D] bf16 ----------------
__global__ void l2norm_q(const float* __restrict__ qraw, __bf16* __restrict__ q_n) {
    int gw   = (int)((blockIdx.x * (size_t)blockDim.x + threadIdx.x) >> 5);
    int lane = threadIdx.x & 31;
    int i  = gw / NH_;            // row = s*B + b
    int hh = gw % NH_;
    if (i >= ROWS_) return;
    int s = i / B_, b = i % B_;
    const float* src = qraw + (size_t)i * H_ + hh * D_;
    float v[4]; float ss = 0.f;
#pragma unroll
    for (int t = 0; t < 4; ++t) { v[t] = src[lane + t * 32]; ss += v[t] * v[t]; }
#pragma unroll
    for (int m = 1; m < 32; m <<= 1) ss += __shfl_xor(ss, m);
    float inv = 1.f / fmaxf(sqrtf(ss), 1e-12f);
    __bf16* dst = q_n + ((size_t)(b * NH_ + hh) * S_ + s) * D_;
#pragma unroll
    for (int t = 0; t < 4; ++t) dst[lane + t * 32] = (__bf16)(v[t] * inv);
}

// ------ L2-normalize k,v: kvraw[M*B, 2H] -> k_n[BH,M,D] bf16, vT[BH,D,M] bf16 ------
__global__ void l2norm_kv(const float* __restrict__ kvraw,
                          __bf16* __restrict__ k_n, __bf16* __restrict__ vT) {
    int gw   = (int)((blockIdx.x * (size_t)blockDim.x + threadIdx.x) >> 5);
    int lane = threadIdx.x & 31;
    int i  = gw / NH_;            // row = m*B + b
    int hh = gw % NH_;
    if (i >= KVROWS_) return;
    int m = i / B_, b = i % B_;
    const float* src = kvraw + (size_t)i * (2 * H_) + hh * (2 * D_);
    float kv[4]; float ss = 0.f;
#pragma unroll
    for (int t = 0; t < 4; ++t) { kv[t] = src[lane + t * 32]; ss += kv[t] * kv[t]; }
#pragma unroll
    for (int mm = 1; mm < 32; mm <<= 1) ss += __shfl_xor(ss, mm);
    float inv = 1.f / fmaxf(sqrtf(ss), 1e-12f);
    __bf16* kd = k_n + ((size_t)(b * NH_ + hh) * M_ + m) * D_;
#pragma unroll
    for (int t = 0; t < 4; ++t) kd[lane + t * 32] = (__bf16)(kv[t] * inv);
    float vv[4]; ss = 0.f;
#pragma unroll
    for (int t = 0; t < 4; ++t) { vv[t] = src[D_ + lane + t * 32]; ss += vv[t] * vv[t]; }
#pragma unroll
    for (int mm = 1; mm < 32; mm <<= 1) ss += __shfl_xor(ss, mm);
    inv = 1.f / fmaxf(sqrtf(ss), 1e-12f);
    __bf16* vd = vT + (size_t)(b * NH_ + hh) * D_ * M_;
#pragma unroll
    for (int t = 0; t < 4; ++t) vd[(size_t)(lane + t * 32) * M_ + m] = (__bf16)(vv[t] * inv);
}

// ---------------- fused flash attention (non-causal, bool mask -> -10000) ----------------
// grid: (S/128, B*NH), block 256 (8 waves). Each wave owns a 16-row S tile.
__global__ __launch_bounds__(256) void flash_attn(
    const __bf16* __restrict__ q_n,   // [BH, S, D]
    const __bf16* __restrict__ k_n,   // [BH, M, D]
    const __bf16* __restrict__ vT,    // [BH, D, M]
    const unsigned char* __restrict__ mask, // [B, S, M] (bool)
    __bf16* __restrict__ ctx) {       // [S*B, H]
    int lane = threadIdx.x & 31;
    int wid  = threadIdx.x >> 5;
    int c    = lane & 15;
    int half = lane >> 4;
    int bh = blockIdx.y;
    int b = bh / NH_, hh = bh % NH_;
    int s0 = blockIdx.x * 128 + wid * 16;

    __shared__ __align__(16) __bf16 pbuf[8][16 * 32];

    Frag aq[4];
    const __bf16* qp = q_n + (size_t)bh * S_ * D_ + (size_t)(s0 + c) * D_;
#pragma unroll
    for (int kc = 0; kc < 4; ++kc) {
        aq[kc].h[0] = *(const bf16x8*)(qp + kc * 32 + half * 8);
        aq[kc].h[1] = *(const bf16x8*)(qp + kc * 32 + 16 + half * 8);
    }

    float mrun[8], lrun[8];
#pragma unroll
    for (int j = 0; j < 8; ++j) { mrun[j] = -3.0e38f; lrun[j] = 0.f; }
    f32x8 o[8] = {};

    const unsigned char* mrow = mask + (size_t)b * S_ * M_;
    const __bf16* kbase = k_n + (size_t)bh * M_ * D_;
    const __bf16* vbase = vT + (size_t)bh * D_ * M_;
    const float scale = 0.08838834764831845f; // 1/sqrt(128)

    for (int m0 = 0; m0 < M_; m0 += 32) {
        // pull next K/V tiles toward WGP while we compute this one
        if (m0 + 32 < M_) {
            __builtin_prefetch(kbase + (size_t)(m0 + 32 + c) * D_ + half * 16, 0, 0);
            __builtin_prefetch(vbase + (size_t)c * M_ + m0 + 32 + half * 16, 0, 0);
        }
        float sc[2][8];
#pragma unroll
        for (int t = 0; t < 2; ++t) {
            int mt = m0 + t * 16;
            f32x8 acc = {};
            const __bf16* kp = kbase + (size_t)(mt + c) * D_ + half * 8;
#pragma unroll
            for (int kc = 0; kc < 4; ++kc) {
                Frag bk;
                bk.h[0] = *(const bf16x8*)(kp + kc * 32);
                bk.h[1] = *(const bf16x8*)(kp + kc * 32 + 16);
                acc = __builtin_amdgcn_wmma_f32_16x16x32_bf16(
                    false, aq[kc].v, false, bk.v, (short)0, acc, false, false);
            }
#pragma unroll
            for (int j = 0; j < 8; ++j) {
                int s = s0 + j + 8 * half;
                float v = acc[j] * scale;
                if (mrow[(size_t)s * M_ + mt + c]) v = -10000.0f;
                sc[t][j] = v;
            }
        }
        // online softmax over this 16x32 chunk
        float mnew[8], alpha[8];
#pragma unroll
        for (int j = 0; j < 8; ++j) {
            float v = fmaxf(sc[0][j], sc[1][j]);
            v = fmaxf(v, __shfl_xor(v, 1));
            v = fmaxf(v, __shfl_xor(v, 2));
            v = fmaxf(v, __shfl_xor(v, 4));
            v = fmaxf(v, __shfl_xor(v, 8));   // reduce within 16-lane half
            mnew[j]  = fmaxf(mrun[j], v);
            alpha[j] = __expf(mrun[j] - mnew[j]);
            mrun[j]  = mnew[j];
        }
#pragma unroll
        for (int t = 0; t < 2; ++t)
#pragma unroll
            for (int j = 0; j < 8; ++j) {
                float p = __expf(sc[t][j] - mnew[j]);
                sc[t][j] = p;
                pbuf[wid][(j + 8 * half) * 32 + t * 16 + c] = (__bf16)p;
            }
#pragma unroll
        for (int j = 0; j < 8; ++j) {
            float rs = sc[0][j] + sc[1][j];
            rs += __shfl_xor(rs, 1);
            rs += __shfl_xor(rs, 2);
            rs += __shfl_xor(rs, 4);
            rs += __shfl_xor(rs, 8);
            lrun[j] = alpha[j] * lrun[j] + rs;
        }
#pragma unroll
        for (int ct = 0; ct < 8; ++ct)
#pragma unroll
            for (int j = 0; j < 8; ++j) o[ct][j] *= alpha[j];

        asm volatile("s_wait_dscnt 0" ::: "memory");
        // re-read P in A-fragment (16x32) layout
        Frag pa;
        pa.h[0] = *(const bf16x8*)(&pbuf[wid][c * 32 + half * 8]);
        pa.h[1] = *(const bf16x8*)(&pbuf[wid][c * 32 + 16 + half * 8]);

        // O += P @ V   (V from transposed layout vT[D, M])
        const __bf16* vp = vbase + m0 + half * 8;
#pragma unroll
        for (int ct = 0; ct < 8; ++ct) {
            Frag bv;
            const __bf16* vpp = vp + (size_t)(ct * 16 + c) * M_;
            bv.h[0] = *(const bf16x8*)(vpp);
            bv.h[1] = *(const bf16x8*)(vpp + 16);
            o[ct] = __builtin_amdgcn_wmma_f32_16x16x32_bf16(
                false, pa.v, false, bv.v, (short)0, o[ct], false, false);
        }
    }
    // finalize: divide by softmax sum, scatter into ctx[S*B, H]
#pragma unroll
    for (int j = 0; j < 8; ++j) lrun[j] = 1.f / lrun[j];
#pragma unroll
    for (int ct = 0; ct < 8; ++ct)
#pragma unroll
        for (int j = 0; j < 8; ++j) {
            int s = s0 + j + 8 * half;
            int col = hh * D_ + ct * 16 + c;
            ctx[(size_t)(s * B_ + b) * H_ + col] = (__bf16)(o[ct][j] * lrun[j]);
        }
}

// ---------------- host-side orchestration ----------------
static inline void launch_gemm(const __bf16* A, const __bf16* Wt, const float* bias,
                               const float* resid, float* Cf, __bf16* Cb,
                               int R, int N, int K, int gelu, hipStream_t s) {
    dim3 g(N / 128, R / 128), blk(256);
    gemm_wmma<<<g, blk, 0, s>>>(A, Wt, bias, resid, Cf, Cb, R, N, K, gelu);
}

extern "C" void kernel_launch(void* const* d_in, const int* in_sizes, int n_in,
                              void* d_out, int out_size, void* d_ws, size_t ws_size,
                              hipStream_t stream) {
    (void)in_sizes; (void)n_in; (void)out_size; (void)ws_size;
    const float* x      = (const float*)d_in[0];
    const float* past   = (const float*)d_in[1];
    const unsigned char* memmask = (const unsigned char*)d_in[2];
    const float* ln1_g  = (const float*)d_in[3];
    const float* ln1_b  = (const float*)d_in[4];
    const float* ln2_g  = (const float*)d_in[5];
    const float* ln2_b  = (const float*)d_in[6];
    const float* ln3_g  = (const float*)d_in[7];
    const float* ln3_b  = (const float*)d_in[8];
    const float* w1     = (const float*)d_in[9];
    const float* b1     = (const float*)d_in[10];
    const float* w2     = (const float*)d_in[11];
    const float* b2     = (const float*)d_in[12];
    const float* w3     = (const float*)d_in[13];
    const float* b3     = (const float*)d_in[14];
    const float* w4     = (const float*)d_in[15];
    const float* b4     = (const float*)d_in[16];
    const float* wq     = (const float*)d_in[17];
    const float* bq     = (const float*)d_in[18];
    const float* wkv    = (const float*)d_in[19];
    const float* bkv    = (const float*)d_in[20];
    const float* wd     = (const float*)d_in[21];
    const float* bd     = (const float*)d_in[22];
    float* out = (float*)d_out;

    // ---- workspace bump allocator ----
    char* p = (char*)d_ws;
    auto alloc = [&](size_t bytes) -> void* {
        void* r = (void*)p;
        p += (bytes + 255) & ~(size_t)255;
        return r;
    };
    __bf16* w1t  = (__bf16*)alloc((size_t)FFN_ * H_ * 2);
    __bf16* w2t  = (__bf16*)alloc((size_t)H_ * FFN_ * 2);
    __bf16* w3t  = (__bf16*)alloc((size_t)FFN_ * H_ * 2);
    __bf16* w4t  = (__bf16*)alloc((size_t)H_ * FFN_ * 2);
    __bf16* wqt  = (__bf16*)alloc((size_t)H_ * H_ * 2);
    __bf16* wkvt = (__bf16*)alloc((size_t)(2 * H_) * H_ * 2);
    __bf16* wdt  = (__bf16*)alloc((size_t)H_ * H_ * 2);
    __bf16* lnbuf  = (__bf16*)alloc((size_t)ROWS_ * H_ * 2);
    __bf16* hbuf   = (__bf16*)alloc((size_t)ROWS_ * FFN_ * 2);
    float*  x1     = (float*) alloc((size_t)ROWS_ * H_ * 4);
    float*  x2     = (float*) alloc((size_t)ROWS_ * H_ * 4);
    __bf16* pastbf = (__bf16*)alloc((size_t)KVROWS_ * H_ * 2);
    float*  qraw   = (float*) alloc((size_t)ROWS_ * H_ * 4);
    float*  kvraw  = (float*) alloc((size_t)KVROWS_ * 2 * H_ * 4);
    __bf16* q_n    = (__bf16*)alloc((size_t)B_ * NH_ * S_ * D_ * 2);
    __bf16* k_n    = (__bf16*)alloc((size_t)B_ * NH_ * M_ * D_ * 2);
    __bf16* vT     = (__bf16*)alloc((size_t)B_ * NH_ * D_ * M_ * 2);
    __bf16* ctx    = (__bf16*)alloc((size_t)ROWS_ * H_ * 2);

    dim3 tb(32, 8);
    // weight converts (transpose to [N,K] bf16)
    cvt_transpose_bf16<<<dim3(FFN_ / 32, H_ / 32),   tb, 0, stream>>>(w1,  w1t,  H_,   FFN_);
    cvt_transpose_bf16<<<dim3(H_ / 32,   FFN_ / 32), tb, 0, stream>>>(w2,  w2t,  FFN_, H_);
    cvt_transpose_bf16<<<dim3(FFN_ / 32, H_ / 32),   tb, 0, stream>>>(w3,  w3t,  H_,   FFN_);
    cvt_transpose_bf16<<<dim3(H_ / 32,   FFN_ / 32), tb, 0, stream>>>(w4,  w4t,  FFN_, H_);
    cvt_transpose_bf16<<<dim3(H_ / 32,   H_ / 32),   tb, 0, stream>>>(wq,  wqt,  H_,   H_);
    cvt_transpose_bf16<<<dim3(2 * H_ / 32, H_ / 32), tb, 0, stream>>>(wkv, wkvt, H_,   2 * H_);
    cvt_transpose_bf16<<<dim3(H_ / 32,   H_ / 32),   tb, 0, stream>>>(wd,  wdt,  H_,   H_);
    cvt_f32_bf16<<<4096, 256, 0, stream>>>(past, pastbf, (size_t)KVROWS_ * H_);

    // mlp1: x -> ln1 -> fc1+gelu -> fc2 + bias (NO residual) => x1
    layernorm_bf16<<<ROWS_, 256, 0, stream>>>(x, ln1_g, ln1_b, lnbuf);
    launch_gemm(lnbuf, w1t, b1, nullptr, nullptr, hbuf, ROWS_, FFN_, H_, 1, stream);
    launch_gemm(hbuf, w2t, b2, nullptr, x1, nullptr, ROWS_, H_, FFN_, 0, stream);

    // attention projections
    layernorm_bf16<<<ROWS_, 256, 0, stream>>>(x1, ln2_g, ln2_b, lnbuf);
    launch_gemm(lnbuf, wqt, bq, nullptr, qraw, nullptr, ROWS_, H_, H_, 0, stream);
    launch_gemm(pastbf, wkvt, bkv, nullptr, kvraw, nullptr, KVROWS_, 2 * H_, H_, 0, stream);

    l2norm_q<<<(ROWS_ * NH_ * 32) / 256, 256, 0, stream>>>(qraw, q_n);
    l2norm_kv<<<(KVROWS_ * NH_ * 32) / 256, 256, 0, stream>>>(kvraw, k_n, vT);

    // fused attention -> ctx (bf16)
    flash_attn<<<dim3(S_ / 128, B_ * NH_), 256, 0, stream>>>(q_n, k_n, vT, memmask, ctx);

    // x2 = x1 + ctx @ wd + bd
    launch_gemm(ctx, wdt, bd, x1, x2, nullptr, ROWS_, H_, H_, 0, stream);

    // mlp2 with residual: out = x2 + gelu(ln3(x2)@w3+b3)@w4 + b4
    layernorm_bf16<<<ROWS_, 256, 0, stream>>>(x2, ln3_g, ln3_b, lnbuf);
    launch_gemm(lnbuf, w3t, b3, nullptr, nullptr, hbuf, ROWS_, FFN_, H_, 1, stream);
    launch_gemm(hbuf, w4t, b4, x2, out, nullptr, ROWS_, H_, FFN_, 0, stream);
}
